// MultiHeadAttention_1159641170485
// MI455X (gfx1250) — matrix operand order repaired and checked by reading, not compiled, not run on hardware
//
#include <hip/hip_runtime.h>
#include <hip/hip_bf16.h>
#include <stdint.h>

typedef __attribute__((ext_vector_type(16))) _Float16 v16h;
typedef __attribute__((ext_vector_type(8)))  float    v8f;

union FragU { v16h h; uint32_t u[8]; };

// A-fragment uint index pattern within a 32-half (16-uint) K chunk.
// 16-bit A 16x32 layout: lanes 0-15 -> uints {0..3, 8..11}; lanes 16-31 -> {4..7, 12..15}
#define UOFF(j, half) ((((j) < 4) ? (j) : (j) + 4) + (half) * 4)

#define HID   1024
#define NHEAD 16
#define DHEAD 64
#define SEQ   2048
#define BT    2
#define RANK  8
#define LORA_SCALING 2.0f   // alpha/rank = 16/8
#define QK_SCALE     0.125f // 64^-0.5

// ---------------------------------------------------------------------------
// CDNA5 async copy: per-lane 16B global -> LDS, tracked with ASYNCcnt.
// LDS byte address from the first VGPR operand; 64-bit global address from the second.
// ---------------------------------------------------------------------------
__device__ __forceinline__ void async_b128(uint32_t lds_byte_off, uint64_t gaddr) {
  asm volatile("global_load_async_to_lds_b128 %0, %1, off"
               :: "v"(lds_byte_off), "v"(gaddr) : "memory");
}
// Non-temporal variant for single-use streams (attn_bias: 537 MB read once;
// keeps the 192 MB L2 for the K/V/Q tiles that are reused 32x).
__device__ __forceinline__ void async_b128_nt(uint32_t lds_byte_off, uint64_t gaddr) {
  asm volatile("global_load_async_to_lds_b128 %0, %1, off th:TH_LOAD_NT"
               :: "v"(lds_byte_off), "v"(gaddr) : "memory");
}
__device__ __forceinline__ void wait_async16() {
  asm volatile("s_wait_asynccnt 0x10" ::: "memory");
}
__device__ __forceinline__ void wait_async0() {
  asm volatile("s_wait_asynccnt 0x0" ::: "memory");
}
__device__ __forceinline__ uint32_t lds_off(const void* p) {
  return (uint32_t)(uintptr_t)p;  // LDS aperture: addr[31:0] is the LDS offset
}

// ---------------------------------------------------------------------------
// Kernel 1: Weff[o][i] = W[o][i] + SCALING * sum_r B[o][r] * A[r][i]  -> half
// ---------------------------------------------------------------------------
__global__ __launch_bounds__(256) void prep_weff_kernel(
    const float* __restrict__ W, const float* __restrict__ A,
    const float* __restrict__ B, _Float16* __restrict__ Weff) {
  int idx = blockIdx.x * 256 + threadIdx.x;   // over 1024*1024
  int o = idx >> 10;
  int i = idx & 1023;
  float acc = W[idx];
  float s = 0.f;
#pragma unroll
  for (int r = 0; r < RANK; ++r) s += B[o * RANK + r] * A[r * HID + i];
  acc += LORA_SCALING * s;
  Weff[idx] = (_Float16)acc;
}

// ---------------------------------------------------------------------------
// Kernel 2: Y[m][o] = (sum_i X[m][i] * Weff[o][i] + bias[o]) * scale  -> half
// Block tile 128x128, 256 threads (8 waves), wave tile 32x64 via WMMA f16.
// ---------------------------------------------------------------------------
__global__ __launch_bounds__(256) void proj_gemm_kernel(
    const float* __restrict__ X, const _Float16* __restrict__ Weff,
    const float* __restrict__ bias, _Float16* __restrict__ Y, float scale) {
  __shared__ __align__(16) _Float16 Xs[128 * 32];
  __shared__ __align__(16) _Float16 Ws[128 * 32];

  const int t    = threadIdx.x;
  const int wave = t >> 5;
  const int lane = t & 31;
  const int lrow = lane & 15;
  const int half = lane >> 4;
  const int wm   = wave & 3;   // 4 waves along M -> 32 rows each
  const int wn   = wave >> 2;  // 2 waves along N -> 64 cols each
  const int m0   = blockIdx.x * 128;
  const int n0   = blockIdx.y * 128;

  const v8f zero8 = {0.f, 0.f, 0.f, 0.f, 0.f, 0.f, 0.f, 0.f};
  v8f acc[2][4];
#pragma unroll
  for (int i = 0; i < 2; ++i)
#pragma unroll
    for (int j = 0; j < 4; ++j) acc[i][j] = zero8;

  const int ldrow = t >> 1;
  const int ldcol = (t & 1) * 16;

  for (int kk = 0; kk < HID; kk += 32) {
    __syncthreads();
    const float* xsrc = X + (size_t)(m0 + ldrow) * HID + kk + ldcol;
    const uint4* wsrc = (const uint4*)(Weff + (size_t)(n0 + ldrow) * HID + kk + ldcol);
    if (kk + 32 < HID) {  // gfx1250 global_prefetch_b8, near-cache locality
      __builtin_prefetch(xsrc + 32, 0, 3);
      __builtin_prefetch((const char*)wsrc + 64, 0, 3);
    }
    // Stage X tile (fp32 -> half) : 128 rows x 32 K
#pragma unroll
    for (int j = 0; j < 16; j += 4) {
      float4 f = *(const float4*)(xsrc + j);
      _Float16* d = &Xs[ldrow * 32 + ldcol + j];
      d[0] = (_Float16)f.x; d[1] = (_Float16)f.y;
      d[2] = (_Float16)f.z; d[3] = (_Float16)f.w;
    }
    // Stage Weff tile (half) : 128 rows(o) x 32 K(i)
    *(uint4*)&Ws[ldrow * 32 + ldcol]     = wsrc[0];
    *(uint4*)&Ws[ldrow * 32 + ldcol + 8] = wsrc[1];
    __syncthreads();

    const uint32_t* xs32 = (const uint32_t*)Xs;
    const uint32_t* ws32 = (const uint32_t*)Ws;

    // Preload ALL fragments into distinct registers so the dscnt waits batch
    // up front and the 8 WMMAs issue back-to-back (round-2 asm showed the
    // compiler recycling one B-frag register set with a wait per WMMA pair).
    FragU a[2], b[4];
#pragma unroll
    for (int ms = 0; ms < 2; ++ms) {
      const int row = wm * 32 + ms * 16 + lrow;   // 16 uints per 32-half row
#pragma unroll
      for (int j = 0; j < 8; ++j) a[ms].u[j] = xs32[row * 16 + UOFF(j, half)];
    }
#pragma unroll
    for (int ns = 0; ns < 4; ++ns) {
      const int orow = wn * 64 + ns * 16 + lrow;
#pragma unroll
      for (int j = 0; j < 8; ++j) b[ns].u[j] = ws32[orow * 16 + UOFF(j, half)];
    }
#pragma unroll
    for (int ns = 0; ns < 4; ++ns)
#pragma unroll
      for (int ms = 0; ms < 2; ++ms)
        acc[ms][ns] = __builtin_amdgcn_wmma_f32_16x16x32_f16(
            false, a[ms].h, false, b[ns].h, (short)0, acc[ms][ns], false, false);
  }

  // Epilogue: bias + scale, store half
#pragma unroll
  for (int ns = 0; ns < 4; ++ns) {
    const int ncol = n0 + wn * 64 + ns * 16 + lrow;
    const float bv = bias[ncol];
#pragma unroll
    for (int ms = 0; ms < 2; ++ms) {
#pragma unroll
      for (int r = 0; r < 8; ++r) {
        const int mrow = m0 + wm * 32 + ms * 16 + r + half * 8;
        Y[(size_t)mrow * HID + ncol] = (_Float16)((acc[ms][ns][r] + bv) * scale);
      }
    }
  }
}

// ---------------------------------------------------------------------------
// Kernel 3: one-time V transpose: Vh[b][s][h*64+d] -> Vt[(b*16+h)*64+d][s]
// so attention can async-copy contiguous V^T rows. 64x64 tile per block.
// ---------------------------------------------------------------------------
__global__ __launch_bounds__(256) void transpose_v_kernel(
    const _Float16* __restrict__ Vh, _Float16* __restrict__ Vt) {
  __shared__ __align__(16) _Float16 T[64][80];  // 160B rows: 16B aligned, conflict-free-ish
  const int t   = threadIdx.x;
  const int s0  = blockIdx.x * 64;
  const int bh  = blockIdx.y;              // b*NHEAD + h
  const int b2  = bh / NHEAD, h2 = bh % NHEAD;
  const int row = t >> 1, seg = (t & 1) * 32;

  const _Float16* src = Vh + ((size_t)b2 * SEQ + s0 + row) * HID + h2 * DHEAD + seg;
#pragma unroll
  for (int j = 0; j < 4; ++j)
    *(uint4*)&T[row][seg + j * 8] = *(const uint4*)(src + j * 8);
  __syncthreads();

  const int drow = t >> 1;
  uint32_t* dst = (uint32_t*)(Vt + ((size_t)bh * DHEAD + drow) * SEQ + s0 + seg);
#pragma unroll
  for (int m = 0; m < 16; ++m) {
    union { uint32_t u; _Float16 e[2]; } p;
    p.e[0] = T[seg + 2 * m][drow];
    p.e[1] = T[seg + 2 * m + 1][drow];
    dst[m] = p.u;
  }
}

// ---------------------------------------------------------------------------
// Kernel 4: flash attention, double-buffered async-to-LDS staging.
// One workgroup per (batch, head, 64 q-rows); 4 waves own 16 q-rows each.
// ---------------------------------------------------------------------------
__global__ __launch_bounds__(128) void attn_kernel(
    const _Float16* __restrict__ Qh, const _Float16* __restrict__ Kh,
    const _Float16* __restrict__ Vt, const float* __restrict__ bias,
    float* __restrict__ out) {
  __shared__ __align__(16) _Float16 Kbuf[2][64 * 64];  // [kcol][d]
  __shared__ __align__(16) _Float16 Vbuf[2][64 * 64];  // [d][kcol]
  __shared__ __align__(16) float    Bbuf[2][64 * 64];  // bias tile [qrow][kcol]
  __shared__ __align__(16) _Float16 Ps[64 * 64];       // [qrow_local][kcol_local]

  const int t    = threadIdx.x;
  const int wave = t >> 5;
  const int lane = t & 31;
  const int lrow = lane & 15;
  const int half = lane >> 4;
  const int q0   = blockIdx.x * 64;
  const int hh   = blockIdx.y;
  const int bb   = blockIdx.z;

  const size_t qkv_base = (size_t)bb * SEQ * HID + (size_t)hh * DHEAD;
  const char* kg = (const char*)(Kh + qkv_base);                        // + kt*2048B rows
  const char* vg = (const char*)(Vt + ((size_t)bb * NHEAD + hh) * DHEAD * SEQ); // rows of 4096B
  const char* bg = (const char*)(bias + ((((size_t)bb * NHEAD + hh) * SEQ) + q0) * SEQ);

  // Stage tile kt into buffer `buf`: 16 async B128 ops per thread.
  auto stage = [&](int kt, int buf) {
    const uint32_t klds = lds_off(&Kbuf[buf][0]);
    const uint32_t vlds = lds_off(&Vbuf[buf][0]);
    const uint32_t blds = lds_off(&Bbuf[buf][0]);
#pragma unroll
    for (int j = 0; j < 4; ++j) {                 // K tile: 64 rows x 128B
      const int i = j * 128 + t, row = i >> 3, part = i & 7;
      async_b128(klds + row * 128 + part * 16,
                 (uint64_t)(uintptr_t)(kg + (size_t)(kt + row) * (HID * 2) + part * 16));
    }
#pragma unroll
    for (int j = 0; j < 4; ++j) {                 // V^T tile: 64 d-rows x 128B
      const int i = j * 128 + t, row = i >> 3, part = i & 7;
      async_b128(vlds + row * 128 + part * 16,
                 (uint64_t)(uintptr_t)(vg + (size_t)row * (SEQ * 2) + kt * 2 + part * 16));
    }
#pragma unroll
    for (int j = 0; j < 8; ++j) {                 // bias tile: 64 q-rows x 256B (NT stream)
      const int i = j * 128 + t, row = i >> 4, part = i & 15;
      async_b128_nt(blds + row * 256 + part * 16,
                    (uint64_t)(uintptr_t)(bg + (size_t)row * (SEQ * 4) + kt * 4 + part * 16));
    }
  };

  // Q A-fragments straight from global (one gather, reused for all 32 k-tiles)
  FragU aq[2];
  {
    const int s = q0 + wave * 16 + lrow;
    const uint32_t* qp = (const uint32_t*)(Qh + qkv_base + (size_t)s * HID);
#pragma unroll
    for (int dc = 0; dc < 2; ++dc)
#pragma unroll
      for (int j = 0; j < 8; ++j) aq[dc].u[j] = qp[dc * 16 + UOFF(j, half)];
  }

  const v8f zero8 = {0.f, 0.f, 0.f, 0.f, 0.f, 0.f, 0.f, 0.f};
  v8f o_acc[4];
  float mrun[8], lsum[8];
#pragma unroll
  for (int nd = 0; nd < 4; ++nd) o_acc[nd] = zero8;
#pragma unroll
  for (int r = 0; r < 8; ++r) { mrun[r] = -1e30f; lsum[r] = 0.f; }

  stage(0, 0);  // prologue

  for (int it = 0; it < SEQ / 64; ++it) {
    const int kt  = it * 64;
    const int buf = it & 1;
    if (it + 1 < SEQ / 64) {
      stage(kt + 64, buf ^ 1);   // prefetch next tile (ASYNCcnt += 16)
      wait_async16();            // current buffer's 16 ops complete
    } else {
      wait_async0();
    }
    __syncthreads();             // all waves' staging of `buf` visible

    // S = Q*K^T  (scale folded into Q); preload both K frags, then 2 WMMAs
    const uint32_t* ks32 = (const uint32_t*)&Kbuf[buf][0];
    v8f sc[4];
#pragma unroll
    for (int n = 0; n < 4; ++n) {
      FragU bk[2];
      const int krow = n * 16 + lrow;              // 32 uints per 64-half row
#pragma unroll
      for (int dc = 0; dc < 2; ++dc)
#pragma unroll
        for (int j = 0; j < 8; ++j) bk[dc].u[j] = ks32[krow * 32 + dc * 16 + UOFF(j, half)];
      sc[n] = zero8;
#pragma unroll
      for (int dc = 0; dc < 2; ++dc)
        sc[n] = __builtin_amdgcn_wmma_f32_16x16x32_f16(
            false, aq[dc].h, false, bk[dc].h, (short)0, sc[n], false, false);
    }
    // Fuse attention bias from the async-staged LDS tile
    const float* bp = &Bbuf[buf][0];
#pragma unroll
    for (int n = 0; n < 4; ++n)
#pragma unroll
      for (int r = 0; r < 8; ++r)
        sc[n][r] += bp[(wave * 16 + r + half * 8) * 64 + n * 16 + lrow];

    // Online softmax: row stats live in 16-lane half-wave groups
    float mx[8];
#pragma unroll
    for (int r = 0; r < 8; ++r)
      mx[r] = fmaxf(fmaxf(sc[0][r], sc[1][r]), fmaxf(sc[2][r], sc[3][r]));
#pragma unroll
    for (int off = 8; off >= 1; off >>= 1)
#pragma unroll
      for (int r = 0; r < 8; ++r) mx[r] = fmaxf(mx[r], __shfl_xor(mx[r], off, 32));

    float alpha[8], rs[8];
#pragma unroll
    for (int r = 0; r < 8; ++r) {
      const float mn = fmaxf(mrun[r], mx[r]);
      alpha[r] = __expf(mrun[r] - mn);
      mrun[r] = mn;
      rs[r] = 0.f;
    }
#pragma unroll
    for (int n = 0; n < 4; ++n)
#pragma unroll
      for (int r = 0; r < 8; ++r) {
        const float p = __expf(sc[n][r] - mrun[r]);
        sc[n][r] = p;
        rs[r] += p;
      }
#pragma unroll
    for (int off = 8; off >= 1; off >>= 1)
#pragma unroll
      for (int r = 0; r < 8; ++r) rs[r] += __shfl_xor(rs[r], off, 32);
#pragma unroll
    for (int r = 0; r < 8; ++r) lsum[r] = lsum[r] * alpha[r] + rs[r];
#pragma unroll
    for (int nd = 0; nd < 4; ++nd)
#pragma unroll
      for (int r = 0; r < 8; ++r) o_acc[nd][r] *= alpha[r];

    // P: C-layout -> A-layout via this wave's private LDS strip (same-wave DS
    // ops are in-order; strip is wave-exclusive, no barrier needed)
#pragma unroll
    for (int n = 0; n < 4; ++n)
#pragma unroll
      for (int r = 0; r < 8; ++r)
        Ps[(wave * 16 + r + half * 8) * 64 + n * 16 + lrow] = (_Float16)sc[n][r];

    FragU pa[2];
    const uint32_t* ps32 = (const uint32_t*)Ps;
#pragma unroll
    for (int kc = 0; kc < 2; ++kc)
#pragma unroll
      for (int j = 0; j < 8; ++j)
        pa[kc].u[j] = ps32[(wave * 16 + lrow) * 32 + kc * 16 + UOFF(j, half)];

    // O += P*V ; preload both V frags per d-tile, then 2 WMMAs
    const uint32_t* vt32 = (const uint32_t*)&Vbuf[buf][0];
#pragma unroll
    for (int nd = 0; nd < 4; ++nd) {
      FragU bv[2];
      const int drow = nd * 16 + lrow;
#pragma unroll
      for (int kc = 0; kc < 2; ++kc)
#pragma unroll
        for (int j = 0; j < 8; ++j) bv[kc].u[j] = vt32[drow * 32 + kc * 16 + UOFF(j, half)];
#pragma unroll
      for (int kc = 0; kc < 2; ++kc)
        o_acc[nd] = __builtin_amdgcn_wmma_f32_16x16x32_f16(
            false, pa[kc].h, false, bv[kc].h, (short)0, o_acc[nd], false, false);
    }
    __syncthreads();  // all waves done reading `buf` before it is restaged
  }

  // Epilogue: normalize and store fp32 out[b][q][h*64+d]
#pragma unroll
  for (int nd = 0; nd < 4; ++nd)
#pragma unroll
    for (int r = 0; r < 8; ++r) {
      const size_t oa = ((size_t)bb * SEQ + (q0 + wave * 16 + r + half * 8)) * HID +
                        hh * DHEAD + nd * 16 + lrow;
      out[oa] = o_acc[nd][r] / lsum[r];
    }
}

// ---------------------------------------------------------------------------
// Launch
// ---------------------------------------------------------------------------
extern "C" void kernel_launch(void* const* d_in, const int* in_sizes, int n_in,
                              void* d_out, int out_size, void* d_ws, size_t ws_size,
                              hipStream_t stream) {
  (void)in_sizes; (void)n_in; (void)out_size; (void)ws_size;

  const float* q  = (const float*)d_in[0];
  const float* k  = (const float*)d_in[1];
  const float* v  = (const float*)d_in[2];
  const float* ab = (const float*)d_in[3];

  const float* W[3]  = {(const float*)d_in[4],  (const float*)d_in[8],  (const float*)d_in[12]};
  const float* bb[3] = {(const float*)d_in[5],  (const float*)d_in[9],  (const float*)d_in[13]};
  const float* A[3]  = {(const float*)d_in[6],  (const float*)d_in[10], (const float*)d_in[14]};
  const float* B[3]  = {(const float*)d_in[7],  (const float*)d_in[11], (const float*)d_in[15]};

  const float* X[3] = {q, k, v};
  const float scales[3] = {QK_SCALE, 1.f, 1.f};

  char* ws = (char*)d_ws;
  const size_t MB = 1024 * 1024;
  _Float16* Weff[3] = {(_Float16*)(ws + 0 * MB), (_Float16*)(ws + 2 * MB),
                       (_Float16*)(ws + 4 * MB)};
  _Float16* Yh[3] = {(_Float16*)(ws + 6 * MB),   // Qh: 4096x1024 half = 8 MB
                     (_Float16*)(ws + 14 * MB),  // Kh
                     (_Float16*)(ws + 22 * MB)}; // Vh
  _Float16* Vt = (_Float16*)(ws + 30 * MB);      // V^T: 8 MB (total 38 MB)

  for (int p = 0; p < 3; ++p)
    prep_weff_kernel<<<dim3((HID * HID) / 256), dim3(256), 0, stream>>>(
        W[p], A[p], B[p], Weff[p]);

  for (int p = 0; p < 3; ++p)
    proj_gemm_kernel<<<dim3((BT * SEQ) / 128, HID / 128), dim3(256), 0, stream>>>(
        X[p], Weff[p], bb[p], Yh[p], scales[p]);

  transpose_v_kernel<<<dim3(SEQ / 64, BT * NHEAD), dim3(256), 0, stream>>>(Yh[2], Vt);

  attn_kernel<<<dim3(SEQ / 64, NHEAD, BT), dim3(128), 0, stream>>>(
      Yh[0], Yh[1], Vt, ab, (float*)d_out);
}
